// MemoryMatrixBlock_9131100472062
// MI455X (gfx1250) — matrix operand order repaired and checked by reading, not compiled, not run on hardware
//
#include <hip/hip_runtime.h>
#include <math.h>

// Problem constants (from reference)
#define GG   9
#define BB   64
#define CC   256
#define WW   24
#define SS   1024
#define DD   16384
#define THRES 0.0025f
#define EPSF  1e-12f

typedef __attribute__((ext_vector_type(16))) __bf16 v16bf;
typedef __attribute__((ext_vector_type(2)))  __bf16 v2bf;
typedef __attribute__((ext_vector_type(8)))  float  v8f;

union FragU { uint4 q[2]; unsigned u[8]; v16bf v; };
union PairU { v2bf v; unsigned u; };

// Split a pair of fp32 into packed bf16 hi (RZ truncation, one v_perm_b32)
// and packed bf16 lo (residual, v_pk_add + v_cvt_pk_bf16_f32).
static __device__ __forceinline__ void split2(float a0, float a1,
                                              unsigned& hpk, unsigned& lpk) {
  unsigned u0 = __float_as_uint(a0);
  unsigned u1 = __float_as_uint(a1);
  hpk = __builtin_amdgcn_perm(u1, u0, 0x07060302u);   // {hi16(a1), hi16(a0)}
  float r0 = a0 - __uint_as_float(u0 & 0xffff0000u);
  float r1 = a1 - __uint_as_float(u1 & 0xffff0000u);
  PairU p;
  p.v[0] = (__bf16)r0;
  p.v[1] = (__bf16)r1;
  lpk = p.u;
}

// Split 16 fp32 (register path, for the B stream) into hi/lo v16bf fragments.
static __device__ __forceinline__ void split16(const float* __restrict__ f,
                                               v16bf& hi, v16bf& lo) {
  FragU H, L;
#pragma unroll
  for (int i = 0; i < 8; ++i) split2(f[2 * i], f[2 * i + 1], H.u[i], L.u[i]);
  hi = H.v;
  lo = L.v;
}

// ---------------------------------------------------------------------------
// Kernel 0: window partition x[B,C,W,H] -> xw[B,G,D], d = (wi*8+hi)*256+ch.
// Reads x HBM-coalesced; scattered 4B writes absorbed by the 192MB L2.
// xw is staged in d_out (same element count); overwritten later by k_out.
// ---------------------------------------------------------------------------
__global__ void k_pack(const float* __restrict__ x, float* __restrict__ xw) {
  int idx = blockIdx.x * 256 + threadIdx.x;          // 9,437,184 total, exact
  int h   = idx % 24;
  int tmp = idx / 24;
  int w   = tmp % 24;  tmp /= 24;
  int ch  = tmp % 256;
  int b   = tmp / 256;
  int g   = (w >> 3) * 3 + (h >> 3);
  int d   = (((w & 7) << 3) + (h & 7)) * 256 + ch;
  xw[((size_t)b * GG + g) * DD + d] = x[idx];
}

// LDS A-tile planes: 64 rows x 16 dwords (32 bf16), stride 20 dwords.
// Row layout is contiguous so lane-half kh reads dwords [kh*8..kh*8+7],
// i.e. K = 16*kh + s, matching the B fragment order:
//   dwords 0..3  = K-pairs 0..3   (K 0..7)
//   dwords 4..7  = K-pairs 4..7   (K 8..15)
//   dwords 8..11 = K-pairs 8..11  (K 16..23)
//   dwords 12..15= K-pairs 12..15 (K 24..31)
#define ROWDW 20

// Convert 32 fp32 (one A row for one K-step) and store hi/lo planes in K order.
static __device__ __forceinline__ void stage_row(const float4* __restrict__ src,
                                                 unsigned* __restrict__ Ah,
                                                 unsigned* __restrict__ Al,
                                                 int row) {
  float4 f[8];
#pragma unroll
  for (int j = 0; j < 8; ++j) f[j] = src[j];
  unsigned hu[16], lu[16];
#pragma unroll
  for (int q = 0; q < 8; ++q) {
    split2(f[q].x, f[q].y, hu[2 * q],     lu[2 * q]);
    split2(f[q].z, f[q].w, hu[2 * q + 1], lu[2 * q + 1]);
  }
  uint4* hr = (uint4*)&Ah[row * ROWDW];
  uint4* lr = (uint4*)&Al[row * ROWDW];
  hr[0] = make_uint4(hu[0],  hu[1],  hu[2],  hu[3]);
  hr[1] = make_uint4(hu[4],  hu[5],  hu[6],  hu[7]);
  hr[2] = make_uint4(hu[8],  hu[9],  hu[10], hu[11]);
  hr[3] = make_uint4(hu[12], hu[13], hu[14], hu[15]);
  lr[0] = make_uint4(lu[0],  lu[1],  lu[2],  lu[3]);
  lr[1] = make_uint4(lu[4],  lu[5],  lu[6],  lu[7]);
  lr[2] = make_uint4(lu[8],  lu[9],  lu[10], lu[11]);
  lr[3] = make_uint4(lu[12], lu[13], lu[14], lu[15]);
}

static __device__ __forceinline__ v16bf frag_load(const unsigned* __restrict__ P,
                                                  int row, int kh) {
  const uint4* rp = (const uint4*)&P[row * ROWDW + kh * 8];
  FragU U;
  U.q[0] = rp[0];
  U.q[1] = rp[1];
  return U.v;
}

// ---------------------------------------------------------------------------
// Kernel 1: att[g,b,s] = shrink( xw[b,g,:] . mem[g,s,:] )
// Tile: M=64(B) x N=32(S), K=16384 loop. Block = 64 thr (2 waves, wave32),
// each wave owns a 16-slot strip; 4 M-tiles => 4 accumulators/wave.
// A is split hi/lo ONCE per WG per K-step into LDS; 3x bf16 WMMA per tile.
// ---------------------------------------------------------------------------
__global__ void __launch_bounds__(64)
k_att(const float* __restrict__ xw, const float* __restrict__ mem,
      float* __restrict__ att) {
  __shared__ unsigned Ah[64 * ROWDW];
  __shared__ unsigned Al[64 * ROWDW];
  const int g    = blockIdx.y;
  const int s0   = blockIdx.x * 32;
  const int t    = threadIdx.x;
  const int lane = t & 31;
  const int wv   = t >> 5;
  const int l15  = lane & 15;
  const int kh   = lane >> 4;

  // B stream: each lane owns one slot row; lanes L and L+16 cover 128B/row.
  const float* Bp = mem + ((size_t)g * SS + (s0 + wv * 16 + l15)) * DD;
  // A staging: thread t loads + splits xw row b=t (32 floats per K-step).
  const float* Ap = xw + ((size_t)t * GG + g) * DD;

  v8f acc[4] = {};

  for (int it = 0; it < 512; ++it) {
    const int k0 = it * 32;
    // Prefetch the B stream ~8 K-steps (1KB) ahead.
    __builtin_prefetch(Bp + k0 + 256, 0, 0);

    float bsrc[16];
    const float4* bp4 = (const float4*)(Bp + k0 + kh * 16);
#pragma unroll
    for (int j = 0; j < 4; ++j) ((float4*)bsrc)[j] = bp4[j];

    __syncthreads();
    stage_row((const float4*)(Ap + k0), Ah, Al, t);
    __syncthreads();

    v16bf bh, bl;
    split16(bsrc, bh, bl);

#pragma unroll
    for (int mt = 0; mt < 4; ++mt) {
      const int row = mt * 16 + l15;
      v16bf ah = frag_load(Ah, row, kh);
      v16bf al = frag_load(Al, row, kh);
      acc[mt] = __builtin_amdgcn_wmma_f32_16x16x32_bf16(false, al, false, bh,
                                                        (short)0, acc[mt], false, false);
      acc[mt] = __builtin_amdgcn_wmma_f32_16x16x32_bf16(false, ah, false, bl,
                                                        (short)0, acc[mt], false, false);
      acc[mt] = __builtin_amdgcn_wmma_f32_16x16x32_bf16(false, ah, false, bh,
                                                        (short)0, acc[mt], false, false);
    }
  }

  // Epilogue: hard_shrink_relu, store raw (unnormalized) att.
  const int scol = s0 + wv * 16 + l15;
#pragma unroll
  for (int mt = 0; mt < 4; ++mt) {
#pragma unroll
    for (int r = 0; r < 8; ++r) {
      const int brow = mt * 16 + kh * 8 + r;         // C/D layout: M=r+8*(lane>>4)
      float a  = acc[mt][r];
      float tt = a - THRES;
      float p  = fmaxf(tt, 0.f);
      float sv = p * a / (fabsf(tt) + EPSF);
      att[((size_t)g * BB + brow) * SS + scol] = sv;
    }
  }
}

// ---------------------------------------------------------------------------
// Kernel 1b: inv_l1[g*64+b] = 1 / max(sum_s |att|, eps). One wave per row.
// ---------------------------------------------------------------------------
__global__ void k_l1(const float* __restrict__ att, float* __restrict__ inv) {
  const int lane = threadIdx.x & 31;
  const int row  = blockIdx.x * (blockDim.x >> 5) + (threadIdx.x >> 5);  // 0..575
  const float* ar = att + (size_t)row * SS;
  float s = 0.f;
  for (int i = lane; i < SS; i += 32) s += fabsf(ar[i]);
#pragma unroll
  for (int off = 16; off; off >>= 1) s += __shfl_xor(s, off, 32);
  if (lane == 0) inv[row] = 1.0f / fmaxf(s, EPSF);
}

// ---------------------------------------------------------------------------
// Kernel 2: out[b,g,:] = (att[g,b,:]/l1) . mem[g], scattered back to y[B,C,W,H].
// Tile: M=64(B) x N=32(D cols), K=1024 loop. Normalization deferred to the
// output scale (per-(b,g) scalar).
// ---------------------------------------------------------------------------
__global__ void __launch_bounds__(64)
k_out(const float* __restrict__ att, const float* __restrict__ mem,
      const float* __restrict__ inv, float* __restrict__ y) {
  __shared__ unsigned Ah[64 * ROWDW];
  __shared__ unsigned Al[64 * ROWDW];
  const int g    = blockIdx.y;
  const int n0   = blockIdx.x * 32;                  // D-column tile (one pixel, 32 ch)
  const int t    = threadIdx.x;
  const int lane = t & 31;
  const int wv   = t >> 5;
  const int l15  = lane & 15;
  const int kh   = lane >> 4;

  const int pix = n0 >> 8;                           // wi*8+hi
  const int ch  = (n0 & 255) + wv * 16 + l15;
  const int wc  = (g / 3) * 8 + (pix >> 3);
  const int hc  = (g % 3) * 8 + (pix & 7);

  const int ncol = n0 + wv * 16 + l15;
  const float* Bp = mem + (size_t)g * SS * DD + ncol;
  const float* Ap = att + ((size_t)g * BB + t) * SS; // thread t stages row b=t

  v8f acc[4] = {};

  for (int it = 0; it < 32; ++it) {
    const int k0 = it * 32;

    // B fragment: 16 dword loads; lanes 0..15 are consecutive columns -> 64B lines.
    float bsrc[16];
#pragma unroll
    for (int e = 0; e < 16; ++e)
      bsrc[e] = Bp[(size_t)(k0 + kh * 16 + e) * DD];

    __syncthreads();
    stage_row((const float4*)(Ap + k0), Ah, Al, t);
    __syncthreads();

    v16bf bh, bl;
    split16(bsrc, bh, bl);

#pragma unroll
    for (int mt = 0; mt < 4; ++mt) {
      const int row = mt * 16 + l15;
      v16bf ah = frag_load(Ah, row, kh);
      v16bf al = frag_load(Al, row, kh);
      acc[mt] = __builtin_amdgcn_wmma_f32_16x16x32_bf16(false, al, false, bh,
                                                        (short)0, acc[mt], false, false);
      acc[mt] = __builtin_amdgcn_wmma_f32_16x16x32_bf16(false, ah, false, bl,
                                                        (short)0, acc[mt], false, false);
      acc[mt] = __builtin_amdgcn_wmma_f32_16x16x32_bf16(false, ah, false, bh,
                                                        (short)0, acc[mt], false, false);
    }
  }

  // Scale by 1/l1 and scatter to y[B,C,W,H] (4B scatter, absorbed by L2).
#pragma unroll
  for (int mt = 0; mt < 4; ++mt) {
#pragma unroll
    for (int r = 0; r < 8; ++r) {
      const int b = mt * 16 + kh * 8 + r;
      float v = acc[mt][r] * inv[g * BB + b];
      y[(((size_t)b * CC + ch) * WW + wc) * WW + hc] = v;
    }
  }
}

// ---------------------------------------------------------------------------
extern "C" void kernel_launch(void* const* d_in, const int* in_sizes, int n_in,
                              void* d_out, int out_size, void* d_ws, size_t ws_size,
                              hipStream_t stream) {
  const float* x   = (const float*)d_in[0];   // [64,256,24,24] fp32
  const float* mem = (const float*)d_in[1];   // [9,1024,16384] fp32
  float* y   = (float*)d_out;                 // [64,256,24,24] fp32
  float* att = (float*)d_ws;                  // [9,64,1024] fp32 (2.36 MB)
  float* inv = att + (size_t)GG * BB * SS;    // [9*64] fp32
  float* xw  = y;                             // stage xw in d_out, k_out overwrites

  k_pack<<<dim3(36864), dim3(256), 0, stream>>>(x, xw);
  k_att <<<dim3(32, 9), dim3(64),  0, stream>>>(xw, mem, att);
  k_l1  <<<dim3(72),    dim3(256), 0, stream>>>(att, inv);
  k_out <<<dim3(512, 9), dim3(64), 0, stream>>>(att, mem, inv, y);
}
